// MultiHeadAttention_74397423501412
// MI455X (gfx1250) — matrix operand order repaired and checked by reading, not compiled, not run on hardware
//
#include <hip/hip_runtime.h>

// ---------------------------------------------------------------------------
// MI455X / gfx1250 fused multi-head attention (bf16 WMMA + async-LDS DMA)
//   D_MODEL=1024, D_K=D_V=64, H=16, B=4, N=2048
// Pipeline:
//   1) convert Q/K/V to bf16; convert+transpose weights to bf16 K-major
//   2) unified GEMM (async double-buffered LDS staging, WMMA bf16 core)
//      -> qh [B,H,N,64] (log2e/sqrt(dk) folded), kh [B,H,N,64], vT [B,H,64,N]
//   3) flash attention (online softmax, per-wave causal bound, L2-resident K/V)
//   4) same GEMM kernel for the output projection (f32 out + bias)
// ---------------------------------------------------------------------------

#define D_MODEL 1024
#define SEQ_N   2048
#define NHEAD   16
#define NBATCH  4
#define DHEAD   64

typedef __bf16 bf16;
typedef __attribute__((ext_vector_type(16))) __bf16 v16bf;
typedef __attribute__((ext_vector_type(8)))  __bf16 v8bf;
typedef __attribute__((ext_vector_type(8)))  float  v8f;

union bf16x16 { v16bf v; v8bf h[2]; };

__device__ __forceinline__ v8f wmma_bf16(const bf16x16& a, const bf16x16& b, v8f c) {
  return __builtin_amdgcn_wmma_f32_16x16x32_bf16(false, a.v, false, b.v,
                                                 (short)0, c, false, false);
}

// Async DMA: 16 bytes global -> LDS per lane, tracked by ASYNCcnt.
__device__ __forceinline__ void async_copy16(unsigned lds_off, const bf16* gptr) {
  asm volatile("global_load_async_to_lds_b128 %0, %1, off"
               :: "v"(lds_off), "v"(gptr) : "memory");
}
__device__ __forceinline__ void wait_async0() {
  asm volatile("s_wait_asynccnt 0" ::: "memory");
}

// ---------------------------------------------------------------------------
// Pre-convert kernels (one pass; makes all GEMM staging pure bf16 DMA)
// ---------------------------------------------------------------------------
__global__ __launch_bounds__(256)
void cvt_f32_bf16_kernel(const float* __restrict__ in, bf16* __restrict__ out, int n) {
  int i = blockIdx.x * 256 + threadIdx.x;
  if (i < n) out[i] = (bf16)in[i];
}

// in[g][k][n] (f32) -> out[g][n][k] (bf16, K-contiguous for B-fragment loads)
__global__ __launch_bounds__(256)
void cvt_transpose_kernel(const float* __restrict__ in, bf16* __restrict__ out,
                          int K, int Ncol, int total) {
  int i = blockIdx.x * 256 + threadIdx.x;
  if (i >= total) return;
  int n = i % Ncol;
  int k = (i / Ncol) % K;
  int g = i / (Ncol * K);
  out[(g * Ncol + n) * K + k] = (bf16)in[i];
}

// ---------------------------------------------------------------------------
// Unified GEMM: C(128x64 per block) = X(8192x1024 bf16) * WT-tile + bias
//   WT is [cols][K] bf16 (K-contiguous); blockIdx.y selects a 64-col tile
//   mode 0: outb bf16 [B,H,N,64]   (q/k; q pre-scaled by log2e/sqrt(dk))
//   mode 1: outb bf16 [B,H,64,N]   (v transposed)
//   mode 2: outf f32  [B*N,1024]   (final output projection)
// Async double-buffered LDS staging: DMA next tile while WMMAs run.
// ---------------------------------------------------------------------------
__global__ __launch_bounds__(256)
void gemm_kernel(const bf16* __restrict__ X, const bf16* __restrict__ WT,
                 const float* __restrict__ bias, bf16* __restrict__ outb,
                 float* __restrict__ outf, int mode, float scale) {
  __shared__ bf16 Xs[2][128][32];   // 2 x 8KB
  __shared__ bf16 Ws[2][64][32];    // 2 x 4KB

  const int tid  = threadIdx.x;
  const int wave = tid >> 5;
  const int lane = tid & 31;
  const int half = lane >> 4;
  const int l16  = lane & 15;
  const int row0 = blockIdx.x * 128;
  const int cb   = blockIdx.y;                 // 64-column tile
  const bf16* Wb = WT + cb * 64 * D_MODEL;

  // Stage one 128x32 X tile + 64x32 W tile into buffer `buf` via async DMA.
  auto stage = [&](int kk, int buf) {
    // X tile: 512 x 16B chunks, 2 per thread (4 chunks per 32-elem row)
#pragma unroll
    for (int i = 0; i < 2; ++i) {
      int e = i * 256 + tid;
      int r = e >> 2, c = e & 3;
      async_copy16((unsigned)(size_t)&Xs[buf][r][c * 8],
                   X + (size_t)(row0 + r) * D_MODEL + kk + c * 8);
    }
    // W tile: 256 x 16B chunks, 1 per thread
    {
      int r = tid >> 2, c = tid & 3;
      async_copy16((unsigned)(size_t)&Ws[buf][r][c * 8],
                   Wb + (size_t)r * D_MODEL + kk + c * 8);
    }
  };

  v8f acc[4] = {};

  stage(0, 0);
  wait_async0();
  __syncthreads();

  for (int kk = 0; kk < D_MODEL; kk += 32) {
    const int buf = (kk >> 5) & 1;
    if (kk + 32 < D_MODEL) stage(kk + 32, buf ^ 1);   // overlap DMA with math

    // A fragment: 16 rows x 32 K (lane=row; K {0-7,16-23}/{8-15,24-31})
    bf16x16 a;
    {
      int r  = wave * 16 + l16;
      int kb = half ? 8 : 0;
      a.h[0] = *(const v8bf*)&Xs[buf][r][kb];
      a.h[1] = *(const v8bf*)&Xs[buf][r][kb + 16];
    }
#pragma unroll
    for (int nt = 0; nt < 4; ++nt) {
      // B fragment: 32 K x 16 cols (lane=col; K {0-7,16-23}/{8-15,24-31})
      bf16x16 bm;
      int n  = nt * 16 + l16;
      int kb = half ? 8 : 0;
      bm.h[0] = *(const v8bf*)&Ws[buf][n][kb];
      bm.h[1] = *(const v8bf*)&Ws[buf][n][kb + 16];
      acc[nt] = wmma_bf16(a, bm, acc[nt]);
    }

    wait_async0();        // next tile's DMA done
    __syncthreads();      // publish LDS writes; all reads of `buf` finished
  }

  // Epilogue (C layout: VGPR i -> row i | i+8 per lane half; lane&15 -> col)
#pragma unroll
  for (int nt = 0; nt < 4; ++nt) {
    int col = nt * 16 + l16;
    float bv = bias[cb * 64 + col];
#pragma unroll
    for (int i = 0; i < 8; ++i) {
      int r = row0 + wave * 16 + i + 8 * half;   // global row in [0, B*N)
      float v = (acc[nt][i] + bv) * scale;
      if (mode == 2) {
        outf[r * D_MODEL + cb * 64 + col] = v;
      } else {
        int b = r >> 11, nidx = r & (SEQ_N - 1);
        int idx = (mode == 0)
            ? ((b * NHEAD + cb) * SEQ_N + nidx) * DHEAD + col
            : ((b * NHEAD + cb) * DHEAD + col) * SEQ_N + nidx;
        outb[idx] = (bf16)v;
      }
    }
  }
}

// ---------------------------------------------------------------------------
// Flash attention: block = 128 query rows of one (b,h); 8 waves x 16 rows.
// Per-wave causal key bound (no block barriers inside the loop; all LDS is
// per-wave). P tile round-trips LDS to re-shape C-layout -> A-fragment.
// ---------------------------------------------------------------------------
__global__ __launch_bounds__(256)
void flash_kernel(const bf16* __restrict__ qh, const bf16* __restrict__ kh,
                  const bf16* __restrict__ vT, bf16* __restrict__ cc,
                  const int* __restrict__ maskp) {
  __shared__ bf16 pl[8][16][32];

  const int tid  = threadIdx.x;
  const int wave = tid >> 5;
  const int lane = tid & 31;
  const int half = lane >> 4;
  const int l16  = lane & 15;
  const int bh   = blockIdx.y;               // b*16 + h
  const int b    = bh >> 4, h = bh & 15;
  const int qbase = blockIdx.x * 128;
  const int wrow  = qbase + wave * 16;

  bf16x16 qa[2];
  {
    const bf16* qp = qh + (bh * SEQ_N + wrow + l16) * DHEAD;
    int kb = half ? 8 : 0;
    qa[0].h[0] = *(const v8bf*)(qp + kb);
    qa[0].h[1] = *(const v8bf*)(qp + kb + 16);
    qa[1].h[0] = *(const v8bf*)(qp + 32 + kb);
    qa[1].h[1] = *(const v8bf*)(qp + 32 + kb + 16);
  }

  v8f o[4] = {};
  float mrow[8], lrow[8];
#pragma unroll
  for (int i = 0; i < 8; ++i) { mrow[i] = -3.0e38f; lrow[i] = 0.0f; }

  const int applymask = maskp[0];
  const int ktiles = applymask ? ((wrow + 16 + 31) >> 5) : (SEQ_N >> 5);

  for (int kt = 0; kt < ktiles; ++kt) {
    const int key0 = kt * 32;

    // Prefetch next key/value tiles (hides first-touch HBM latency; later
    // row-blocks of the same (b,h) hit the 192MB L2 directly).
    if (kt + 1 < ktiles) {
      __builtin_prefetch(kh + (bh * SEQ_N + key0 + 32 + l16) * DHEAD, 0, 1);
      __builtin_prefetch(vT + (bh * DHEAD + l16) * SEQ_N + key0 + 32, 0, 1);
    }

    // ---- S = q * k^T for two 16-key tiles ----
    v8f s[2];
#pragma unroll
    for (int t = 0; t < 2; ++t) {
      bf16x16 bk0, bk1;
      const bf16* kp = kh + (bh * SEQ_N + key0 + t * 16 + l16) * DHEAD;
      int kb = half ? 8 : 0;
      bk0.h[0] = *(const v8bf*)(kp + kb);
      bk0.h[1] = *(const v8bf*)(kp + kb + 16);
      bk1.h[0] = *(const v8bf*)(kp + 32 + kb);
      bk1.h[1] = *(const v8bf*)(kp + 32 + kb + 16);
      v8f z = {};
      z = wmma_bf16(qa[0], bk0, z);
      z = wmma_bf16(qa[1], bk1, z);
      s[t] = z;
    }

    // ---- causal mask + online softmax (base-2 domain) ----
#pragma unroll
    for (int i = 0; i < 8; ++i) {
      const int qr = wrow + i + 8 * half;
      if (applymask) {
        if (key0 + l16 > qr)      s[0][i] = -3.0e38f;
        if (key0 + 16 + l16 > qr) s[1][i] = -3.0e38f;
      }
      float tm = fmaxf(s[0][i], s[1][i]);
#pragma unroll
      for (int d2 = 1; d2 < 16; d2 <<= 1)
        tm = fmaxf(tm, __shfl_xor(tm, d2, 32));
      float mn = fmaxf(mrow[i], tm);
      float sc = exp2f(mrow[i] - mn);
      float p0 = exp2f(s[0][i] - mn);
      float p1 = exp2f(s[1][i] - mn);
      float ps = p0 + p1;
#pragma unroll
      for (int d2 = 1; d2 < 16; d2 <<= 1)
        ps += __shfl_xor(ps, d2, 32);
      lrow[i] = lrow[i] * sc + ps;
      mrow[i] = mn;
      o[0][i] *= sc; o[1][i] *= sc; o[2][i] *= sc; o[3][i] *= sc;
      const int rowi = i + 8 * half;
      pl[wave][rowi][l16]      = (bf16)p0;
      pl[wave][rowi][16 + l16] = (bf16)p1;
    }

    // ---- O += P * V ----
    bf16x16 a;
    {
      int kb = half ? 8 : 0;
      a.h[0] = *(const v8bf*)&pl[wave][l16][kb];
      a.h[1] = *(const v8bf*)&pl[wave][l16][kb + 16];
    }
#pragma unroll
    for (int nt = 0; nt < 4; ++nt) {
      bf16x16 bm;
      const bf16* vp = vT + (bh * DHEAD + nt * 16 + l16) * SEQ_N + key0;
      int kb = half ? 8 : 0;
      bm.h[0] = *(const v8bf*)(vp + kb);
      bm.h[1] = *(const v8bf*)(vp + kb + 16);
      o[nt] = wmma_bf16(a, bm, o[nt]);
    }
  }

  // ---- normalize and write concat [B, N, H*64] bf16 ----
#pragma unroll
  for (int i = 0; i < 8; ++i) {
    float inv = 1.0f / lrow[i];
    int qr = wrow + i + 8 * half;
#pragma unroll
    for (int nt = 0; nt < 4; ++nt) {
      int col = h * DHEAD + nt * 16 + l16;
      cc[(b * SEQ_N + qr) * D_MODEL + col] = (bf16)(o[nt][i] * inv);
    }
  }
}

// ---------------------------------------------------------------------------
extern "C" void kernel_launch(void* const* d_in, const int* in_sizes, int n_in,
                              void* d_out, int out_size, void* d_ws, size_t ws_size,
                              hipStream_t stream) {
  const float* Q  = (const float*)d_in[0];
  const float* K  = (const float*)d_in[1];
  const float* V  = (const float*)d_in[2];
  const float* Wq = (const float*)d_in[3];
  const float* bq = (const float*)d_in[4];
  const float* Wk = (const float*)d_in[5];
  const float* bk = (const float*)d_in[6];
  const float* Wv = (const float*)d_in[7];
  const float* bv = (const float*)d_in[8];
  const float* Wo = (const float*)d_in[9];
  const float* bo = (const float*)d_in[10];
  const int* mask = (const int*)d_in[11];

  const int XSZ = NBATCH * SEQ_N * D_MODEL;          // 8,388,608
  const int QSZ = NBATCH * NHEAD * SEQ_N * DHEAD;    // 8,388,608
  const int WSZ = NHEAD * D_MODEL * DHEAD;           // 1,048,576

  bf16* p   = (bf16*)d_ws;
  bf16* Qb  = p;  p += XSZ;
  bf16* Kb  = p;  p += XSZ;
  bf16* Vb  = p;  p += XSZ;
  bf16* WqT = p;  p += WSZ;
  bf16* WkT = p;  p += WSZ;
  bf16* WvT = p;  p += WSZ;
  bf16* WoT = p;  p += WSZ;
  bf16* qh  = p;  p += QSZ;
  bf16* kh  = p;  p += QSZ;
  bf16* vT  = p;  p += QSZ;
  bf16* cc  = p;  p += QSZ;

  dim3 blk(256);

  // 1) one-pass bf16 conversions
  cvt_f32_bf16_kernel<<<(XSZ + 255) / 256, blk, 0, stream>>>(Q, Qb, XSZ);
  cvt_f32_bf16_kernel<<<(XSZ + 255) / 256, blk, 0, stream>>>(K, Kb, XSZ);
  cvt_f32_bf16_kernel<<<(XSZ + 255) / 256, blk, 0, stream>>>(V, Vb, XSZ);
  cvt_transpose_kernel<<<(WSZ + 255) / 256, blk, 0, stream>>>(Wq, WqT, D_MODEL, DHEAD, WSZ);
  cvt_transpose_kernel<<<(WSZ + 255) / 256, blk, 0, stream>>>(Wk, WkT, D_MODEL, DHEAD, WSZ);
  cvt_transpose_kernel<<<(WSZ + 255) / 256, blk, 0, stream>>>(Wv, WvT, D_MODEL, DHEAD, WSZ);
  cvt_transpose_kernel<<<(WSZ + 255) / 256, blk, 0, stream>>>(Wo, WoT, D_MODEL, D_MODEL, WSZ);

  // 2) per-head projections (fold 1/sqrt(dk)*log2(e) into q)
  const float qscale = 0.125f * 1.44269504088896340736f;
  dim3 gproj(NBATCH * SEQ_N / 128, NHEAD);
  gemm_kernel<<<gproj, blk, 0, stream>>>(Qb, WqT, bq, qh, nullptr, 0, qscale);
  gemm_kernel<<<gproj, blk, 0, stream>>>(Kb, WkT, bk, kh, nullptr, 0, 1.0f);
  gemm_kernel<<<gproj, blk, 0, stream>>>(Vb, WvT, bv, vT, nullptr, 1, 1.0f);

  // 3) flash attention
  flash_kernel<<<dim3(SEQ_N / 128, NBATCH * NHEAD), blk, 0, stream>>>(qh, kh, vT, cc, mask);

  // 4) output projection (f32 out + bias)
  gemm_kernel<<<dim3(NBATCH * SEQ_N / 128, D_MODEL / 64), blk, 0, stream>>>(
      cc, WoT, bo, nullptr, (float*)d_out, 2, 1.0f);
}
